// TreeLSTM_56942676410986
// MI455X (gfx1250) — matrix operand order repaired and checked by reading, not compiled, not run on hardware
//
#include <hip/hip_runtime.h>
#include <hip/hip_bf16.h>
#include <stdint.h>

typedef __attribute__((ext_vector_type(16))) __bf16 v16bf;
typedef __attribute__((ext_vector_type(8)))  float  v8f;

#define B_  64
#define N_  1023
#define H_  512
#define D_  10

__device__ __forceinline__ unsigned short f2bf(float f) {
    unsigned u = __float_as_uint(f);
    unsigned r = (u + 0x7FFFu + ((u >> 16) & 1u)) >> 16;  // round-to-nearest-even
    return (unsigned short)r;
}
__device__ __forceinline__ float sigm(float x) { return 1.0f / (1.0f + __expf(-x)); }

// ---------------- conversion / packing kernels ----------------

__global__ void k_f32_to_bf16(const float* __restrict__ src,
                              unsigned short* __restrict__ dst, long n) {
    long tid = (long)blockIdx.x * blockDim.x + threadIdx.x;
    if (tid < n) dst[tid] = f2bf(src[tid]);
}

// Pack weight (K x Ncols, row-major f32) into WMMA B-fragment order:
// tile t = (k/32)*(Ncols/16) + (col/16); within tile: lane-major, 16 bf16/lane.
// lane l holds col = ct*16 + (l&15), K = kt*32 + (l>>4)*8 + (i&7) + (i>>3)*16  (i=0..15)
__global__ void k_pack_weight(const float* __restrict__ W,
                              unsigned short* __restrict__ out,
                              int K, int Ncols) {
    long total = (long)(K >> 5) * (Ncols >> 4) * 512;
    long tid = (long)blockIdx.x * blockDim.x + threadIdx.x;
    if (tid >= total) return;
    int nCt  = Ncols >> 4;
    long tile = tid >> 9;
    int  idx  = (int)(tid & 511);
    int  lane = idx >> 4;
    int  e    = idx & 15;
    int  kt   = (int)(tile / nCt);
    int  ct   = (int)(tile % nCt);
    int  k    = kt * 32 + (lane >> 4) * 8 + (e & 7) + (e >> 3) * 16;
    int  col  = ct * 16 + (lane & 15);
    out[tid] = f2bf(W[(size_t)k * Ncols + col]);
}

// Gather h_ch rows (children of level d) into a dense bf16 matrix (B*n_d, 2H).
__global__ void k_pack_hch(const float* __restrict__ h,
                           unsigned short* __restrict__ A, int d) {
    int n_d = 1 << d;
    int cstart = (2 << d) - 1;
    long total = (long)B_ * n_d * 2 * H_;
    long tid = (long)blockIdx.x * blockDim.x + threadIdx.x;
    if (tid >= total) return;
    int t = (int)(tid & (2 * H_ - 1));   // 0..1023
    long r = tid >> 10;                  // row in (B*n_d)
    int b = (int)(r >> d);
    int j = (int)(r & (n_d - 1));
    A[tid] = f2bf(h[((size_t)(b * N_ + cstart + 2 * j)) * H_ + t]);
}

// ---------------- WMMA GEMM (32x64 per wave, double-buffered) ----------------
// C row mapping: for global row m in [0,M): b = m >> logNd, j = m & maskNd,
// row address = (b*N_ + startRow + j) * ldc. Dense mode: logNd=30,
// maskNd=(1<<30)-1, startRow=0 -> row = m.
__global__ __launch_bounds__(128)
void k_gemm_bf16(const unsigned short* __restrict__ A,   // (M,K) packed dense bf16
                 const unsigned short* __restrict__ Bp,  // pre-packed B fragments
                 float* __restrict__ C,
                 int K, int Ncols, int ldc,
                 int logNd, int maskNd, int startRow, int beta) {
    const int lane = threadIdx.x & 31;
    const int wave = threadIdx.x >> 5;
    const int colBase = (blockIdx.x * 4 + wave) * 64;
    const int rowBase = blockIdx.y * 32;
    const int nCt = Ncols >> 4;
    const int ct0 = colBase >> 4;

    v8f z = {0.f, 0.f, 0.f, 0.f, 0.f, 0.f, 0.f, 0.f};
    v8f acc[2][4];
#pragma unroll
    for (int rt = 0; rt < 2; ++rt)
#pragma unroll
        for (int t = 0; t < 4; ++t) acc[rt][t] = z;

    union Frag { v16bf bf; uint4 q[2]; };

    // A fragment: lane holds row (lane&15), K chunks kb..kb+7 and kb+16..kb+23
    const unsigned short* aPtr0 =
        A + (size_t)(rowBase + (lane & 15)) * K + (lane >> 4) * 8;
    const unsigned short* aPtr1 = aPtr0 + (size_t)16 * K;
    const unsigned short* bLane = Bp + (size_t)lane * 16;

    auto loadA = [](Frag& f, const unsigned short* p) {
        f.q[0] = *(const uint4*)(p);
        f.q[1] = *(const uint4*)(p + 16);
    };
    auto loadB = [](Frag& f, const unsigned short* p) {
        f.q[0] = *(const uint4*)(p);
        f.q[1] = *(const uint4*)(p + 8);
    };

    Frag a0c, a1c, a0n, a1n, bc[4], bn[4];

    // prologue: k = 0 fragments
    loadA(a0c, aPtr0);
    loadA(a1c, aPtr1);
#pragma unroll
    for (int t = 0; t < 4; ++t)
        loadB(bc[t], bLane + (size_t)(ct0 + t) * 512);

    for (int k = 32; k < K; k += 32) {
        // issue next-step loads first so they overlap the 8 WMMAs below
        const unsigned short* bK = bLane + ((size_t)(k >> 5) * nCt) * 512;
        loadA(a0n, aPtr0 + k);
        loadA(a1n, aPtr1 + k);
#pragma unroll
        for (int t = 0; t < 4; ++t)
            loadB(bn[t], bK + (size_t)(ct0 + t) * 512);

#pragma unroll
        for (int t = 0; t < 4; ++t)
            acc[0][t] = __builtin_amdgcn_wmma_f32_16x16x32_bf16(
                false, a0c.bf, false, bc[t].bf, (short)0, acc[0][t], false, false);
#pragma unroll
        for (int t = 0; t < 4; ++t)
            acc[1][t] = __builtin_amdgcn_wmma_f32_16x16x32_bf16(
                false, a1c.bf, false, bc[t].bf, (short)0, acc[1][t], false, false);

        a0c = a0n;
        a1c = a1n;
#pragma unroll
        for (int t = 0; t < 4; ++t) bc[t] = bn[t];
    }

    // epilogue WMMAs for the last k-step
#pragma unroll
    for (int t = 0; t < 4; ++t)
        acc[0][t] = __builtin_amdgcn_wmma_f32_16x16x32_bf16(
            false, a0c.bf, false, bc[t].bf, (short)0, acc[0][t], false, false);
#pragma unroll
    for (int t = 0; t < 4; ++t)
        acc[1][t] = __builtin_amdgcn_wmma_f32_16x16x32_bf16(
            false, a1c.bf, false, bc[t].bf, (short)0, acc[1][t], false, false);

    // C layout: VGPR j -> M = tileRow + (lane>>4)*8 + j, N = colBase+t*16+(lane&15)
#pragma unroll
    for (int rt = 0; rt < 2; ++rt) {
#pragma unroll
        for (int t = 0; t < 4; ++t) {
            int col = colBase + t * 16 + (lane & 15);
#pragma unroll
            for (int j = 0; j < 8; ++j) {
                int m = rowBase + rt * 16 + (lane >> 4) * 8 + j;
                size_t row = (size_t)((m >> logNd) * N_ + startRow + (m & maskNd));
                size_t addr = row * (size_t)ldc + col;
                float v = acc[rt][t][j];
                if (beta) C[addr] += v; else C[addr] = v;
            }
        }
    }
}

// ---------------- elementwise tree updates ----------------

__global__ void k_leaf(const float* __restrict__ iou, const float* __restrict__ b_iou,
                       float* __restrict__ c, float* __restrict__ h) {
    const int d = D_ - 1, n_d = 1 << d, start = n_d - 1;
    long total = (long)B_ * n_d * H_;
    long tid = (long)blockIdx.x * blockDim.x + threadIdx.x;
    if (tid >= total) return;
    int col = (int)(tid & (H_ - 1));
    long r = tid >> 9;
    int j = (int)(r & (n_d - 1));
    int b = (int)(r >> d);
    size_t irow = (size_t)(b * N_ + start + j) * (3 * H_);
    float iv = iou[irow + col]          + b_iou[col];
    float ov = iou[irow + H_ + col]     + b_iou[H_ + col];
    float uv = iou[irow + 2 * H_ + col] + b_iou[2 * H_ + col];
    float cn = sigm(iv) * tanhf(uv);
    float hn = sigm(ov) * tanhf(cn);
    size_t off = (size_t)(b * N_ + start + j) * H_ + col;
    c[off] = cn;
    h[off] = hn;
}

__global__ void k_internal(int d,
                           const float* __restrict__ iou, const float* __restrict__ b_iou,
                           const float* __restrict__ fpre, const float* __restrict__ b_f,
                           float* __restrict__ c, float* __restrict__ h) {
    int n_d = 1 << d, start = n_d - 1, cstart = (2 << d) - 1;
    long total = (long)B_ * n_d * H_;
    long tid = (long)blockIdx.x * blockDim.x + threadIdx.x;
    if (tid >= total) return;
    int col = (int)(tid & (H_ - 1));
    long r = tid >> 9;                 // row in (B*n_d)
    int j = (int)(r & (n_d - 1));
    int b = (int)(r >> d);
    float f0 = sigm(fpre[(size_t)r * 2 * H_ + col]      + b_f[col]);
    float f1 = sigm(fpre[(size_t)r * 2 * H_ + H_ + col] + b_f[H_ + col]);
    float c0 = c[(size_t)(b * N_ + cstart + 2 * j)     * H_ + col];
    float c1 = c[(size_t)(b * N_ + cstart + 2 * j + 1) * H_ + col];
    float csum = f0 * c0 + f1 * c1;
    size_t irow = (size_t)(b * N_ + start + j) * (3 * H_);
    float iv = iou[irow + col]          + b_iou[col];
    float ov = iou[irow + H_ + col]     + b_iou[H_ + col];
    float uv = iou[irow + 2 * H_ + col] + b_iou[2 * H_ + col];
    float cn = sigm(iv) * tanhf(uv) + csum;
    float hn = sigm(ov) * tanhf(cn);
    size_t off = (size_t)(b * N_ + start + j) * H_ + col;
    c[off] = cn;
    h[off] = hn;
}

// ---------------- host orchestration ----------------

extern "C" void kernel_launch(void* const* d_in, const int* in_sizes, int n_in,
                              void* d_out, int out_size, void* d_ws, size_t ws_size,
                              hipStream_t stream) {
    (void)in_sizes; (void)n_in; (void)out_size; (void)ws_size;
    const float* x     = (const float*)d_in[0];
    const float* W_iou = (const float*)d_in[1];
    const float* U_iou = (const float*)d_in[2];
    const float* b_iou = (const float*)d_in[3];
    const float* U_f   = (const float*)d_in[4];
    const float* b_f   = (const float*)d_in[5];
    float* out = (float*)d_out;

    char* ws = (char*)d_ws;
    size_t off = 0;
    auto carve = [&](size_t bytes) -> void* {
        void* p = ws + off;
        off = (off + bytes + 255) & ~(size_t)255;
        return p;
    };

    const long MN = (long)B_ * N_;                 // 65472 rows
    unsigned short* A_bf   = (unsigned short*)carve((size_t)MN * H_ * 2);          // 67 MB
    float*          iou    = (float*)carve((size_t)MN * 3 * H_ * 4);               // 402 MB
    float*          cbuf   = (float*)carve((size_t)MN * H_ * 4);                   // 134 MB
    float*          h1     = (float*)carve((size_t)MN * H_ * 4);                   // 134 MB
    float*          fpre   = (float*)carve((size_t)B_ * 256 * 2 * H_ * 4);         // 67 MB
    unsigned short* Wp_iou = (unsigned short*)carve((size_t)(H_/32)*(3*H_/16)*512*2);
    unsigned short* Up_iou = (unsigned short*)carve((size_t)(2*H_/32)*(3*H_/16)*512*2);
    unsigned short* Up_f   = (unsigned short*)carve((size_t)(2*H_/32)*(2*H_/16)*512*2);

    const int DENSE_LOG = 30, DENSE_MASK = (1 << 30) - 1;

    // pack weights to bf16 B-fragment layout (once per launch; deterministic)
    {
        long t1 = (long)(H_ / 32) * (3 * H_ / 16) * 512;
        k_pack_weight<<<(t1 + 255) / 256, 256, 0, stream>>>(W_iou, Wp_iou, H_, 3 * H_);
        long t2 = (long)(2 * H_ / 32) * (3 * H_ / 16) * 512;
        k_pack_weight<<<(t2 + 255) / 256, 256, 0, stream>>>(U_iou, Up_iou, 2 * H_, 3 * H_);
        long t3 = (long)(2 * H_ / 32) * (2 * H_ / 16) * 512;
        k_pack_weight<<<(t3 + 255) / 256, 256, 0, stream>>>(U_f, Up_f, 2 * H_, 2 * H_);
    }

    for (int pass = 0; pass < 2; ++pass) {
        const float* xin = (pass == 0) ? x : h1;
        float* h = (pass == 0) ? h1 : out;

        // x (or h1) -> bf16 dense A
        long nconv = MN * H_;
        k_f32_to_bf16<<<(nconv + 255) / 256, 256, 0, stream>>>(xin, A_bf, nconv);

        // iou_all = A @ W_iou   (M=65472, K=512, N=1536) dense
        dim3 g1(3 * H_ / 256, (unsigned)(MN / 32));
        k_gemm_bf16<<<g1, 128, 0, stream>>>(A_bf, Wp_iou, iou, H_, 3 * H_, 3 * H_,
                                            DENSE_LOG, DENSE_MASK, 0, 0);

        for (int d = D_ - 1; d >= 0; --d) {
            int n_d = 1 << d;
            long nEl = (long)B_ * n_d * H_;
            if (d == D_ - 1) {
                k_leaf<<<(nEl + 255) / 256, 256, 0, stream>>>(iou, b_iou, cbuf, h);
            } else {
                long nPack = (long)B_ * n_d * 2 * H_;
                k_pack_hch<<<(nPack + 255) / 256, 256, 0, stream>>>(h, A_bf, d);
                int M = B_ * n_d;
                // fpre = h_ch @ U_f   (dense scratch, beta=0)
                dim3 g2(2 * H_ / 256, (unsigned)(M / 32));
                k_gemm_bf16<<<g2, 128, 0, stream>>>(A_bf, Up_f, fpre, 2 * H_, 2 * H_,
                                                    2 * H_, DENSE_LOG, DENSE_MASK, 0, 0);
                // iou_all[:, start:start+n_d] += h_ch @ U_iou  (tree rows, beta=1)
                dim3 g3(3 * H_ / 256, (unsigned)(M / 32));
                k_gemm_bf16<<<g3, 128, 0, stream>>>(A_bf, Up_iou, iou, 2 * H_, 3 * H_,
                                                    3 * H_, d, n_d - 1, n_d - 1, 1);
                k_internal<<<(nEl + 255) / 256, 256, 0, stream>>>(d, iou, b_iou, fpre,
                                                                  b_f, cbuf, h);
            }
        }
    }
}
